// EnhancedHierarchicalMoE_37864431681662
// MI455X (gfx1250) — compile-verified, matrix operand order checked by reading, not run on hardware
//
#include <hip/hip_runtime.h>
#include <cstdint>
#include <cstddef>

// ---------------- problem constants (from reference setup_inputs) ------------
#define NTOK  2048      // B*S
#define DIM   1024      // D
#define HID   4096      // H
#define NEXP  16        // E
#define NGRP  4         // G
#define EPG   4         // experts per group
#define CHALF 512       // D/2 (confidence hidden)
#define TM    32        // tokens per FFN block tile
#define HC    64        // H chunk per FFN iteration

typedef __attribute__((ext_vector_type(16))) __bf16        v16bf;
typedef __attribute__((ext_vector_type(8)))  float         v8f;
typedef __attribute__((ext_vector_type(4)))  unsigned int  v4u;
typedef __attribute__((ext_vector_type(8)))  int           v8i;
typedef __attribute__((ext_vector_type(4)))  int           v4i;

// ---------------- helpers ----------------------------------------------------
__device__ __forceinline__ float gelu_exact(float v) {
  return 0.5f * v * (1.0f + erff(v * 0.70710678118654752f));
}

__device__ __forceinline__ float wave_sum(float v) {
#pragma unroll
  for (int o = 16; o > 0; o >>= 1) v += __shfl_xor(v, o, 32);
  return v;
}

// A-fragment: 16x32 bf16 tile of a row-major fp32 matrix; rowp = this lane's
// row pointer. ISA layout: lanes 0-15 hold K=0..7,16..23; lanes 16-31 hold
// K=8..15,24..31 (ksel = (lane>>4)*8).
__device__ __forceinline__ v16bf load_a_frag(const float* rowp, int kb, int ksel) {
  v16bf f;
  const float* p = rowp + kb + ksel;
#pragma unroll
  for (int j = 0; j < 8; ++j) { f[j] = (__bf16)p[j]; f[8 + j] = (__bf16)p[16 + j]; }
  return f;
}

// B-fragment: 32x16 bf16 tile of row-major fp32 W[K][ldw]; col fixed per lane
// (n = nb + (lane&15)), K pattern mirrors the A layout. Lane-adjacent columns
// make each per-K access a contiguous 64B row segment (coalesced).
__device__ __forceinline__ v16bf load_b_frag(const float* W, int ldw, int kb,
                                             int ksel, int col) {
  v16bf f;
  const float* p = W + (size_t)(kb + ksel) * ldw + col;
#pragma unroll
  for (int j = 0; j < 8; ++j) {
    f[j]     = (__bf16)p[(size_t)j * ldw];
    f[8 + j] = (__bf16)p[(size_t)(16 + j) * ldw];
  }
  return f;
}

// ---------------- kernel 0: zero ---------------------------------------------
__global__ void zero_f32(float* __restrict__ p, int n) {
  int i = blockIdx.x * blockDim.x + threadIdx.x;
  if (i < n) p[i] = 0.0f;
}

// ---------------- kernel 1: C = GELU(x @ Wc1 + bc1)  [NTOK x CHALF] ----------
__global__ __launch_bounds__(256) void conf_hidden_kernel(
    const float* __restrict__ x, const float* __restrict__ Wc1,
    const float* __restrict__ bc1, float* __restrict__ C) {
  const int tid = threadIdx.x, wave = tid >> 5, lane = tid & 31;
  const int lm = lane & 15, ksel = (lane >> 4) * 8;   // ksel doubles as C/D M-offset
  const int n0 = blockIdx.x * 32, c0 = blockIdx.y * 64;
  const int rt = wave & 1, ct = wave >> 1;            // 2 row-tiles x 4 col-tiles
  const float* arow = x + (size_t)(n0 + rt * 16 + lm) * DIM;
  const int col = c0 + ct * 16 + lm;

  v8f acc;
#pragma unroll
  for (int i = 0; i < 8; ++i) acc[i] = 0.0f;

  for (int kb = 0; kb < DIM; kb += 32) {
    v16bf a = load_a_frag(arow, kb, ksel);
    v16bf b = load_b_frag(Wc1, CHALF, kb, ksel, col);
    acc = __builtin_amdgcn_wmma_f32_16x16x32_bf16(false, a, false, b,
                                                  (short)0, acc, false, false);
  }
  const float bias = bc1[col];
#pragma unroll
  for (int i = 0; i < 8; ++i) {
    int row = n0 + rt * 16 + i + ksel;
    C[(size_t)row * CHALF + col] = gelu_exact(acc[i] + bias);
  }
}

// ---------------- kernel 2: conf = sigmoid(C @ Wc2 + bc2) --------------------
__global__ __launch_bounds__(256) void conf_out_kernel(
    const float* __restrict__ C, const float* __restrict__ Wc2,
    const float* __restrict__ bc2, float* __restrict__ conf) {
  const int wave = threadIdx.x >> 5, lane = threadIdx.x & 31;
  const int n = blockIdx.x * 8 + wave;
  float acc = 0.0f;
#pragma unroll
  for (int k = 0; k < CHALF / 32; ++k) {
    int i = lane + 32 * k;
    acc += C[(size_t)n * CHALF + i] * Wc2[i];
  }
  acc = wave_sum(acc) + bc2[0];
  if (lane == 0) conf[n] = 1.0f / (1.0f + expf(-acc));
}

// ---------------- kernel 3: hierarchical router -> cw[N,16] ------------------
__global__ __launch_bounds__(256) void router_kernel(
    const float* __restrict__ x, const float* __restrict__ Wg,
    const float* __restrict__ bg, const float* __restrict__ Wer,
    const float* __restrict__ ber, const float* __restrict__ conf,
    float* __restrict__ cw) {
  const int wave = threadIdx.x >> 5, lane = threadIdx.x & 31;
  const int n = blockIdx.x * 8 + wave;

  float xv[32];
#pragma unroll
  for (int k = 0; k < 32; ++k) xv[k] = x[(size_t)n * DIM + lane + 32 * k];

  float glog[NGRP];
#pragma unroll
  for (int g = 0; g < NGRP; ++g) {
    float a = 0.0f;
#pragma unroll
    for (int k = 0; k < 32; ++k)
      a += xv[k] * Wg[(size_t)(lane + 32 * k) * NGRP + g];
    glog[g] = wave_sum(a) + bg[g];
  }
  float elog[NGRP][EPG];
  for (int g = 0; g < NGRP; ++g)
    for (int e = 0; e < EPG; ++e) {
      float a = 0.0f;
#pragma unroll
      for (int k = 0; k < 32; ++k)
        a += xv[k] * Wer[(size_t)g * DIM * EPG + (size_t)(lane + 32 * k) * EPG + e];
      elog[g][e] = wave_sum(a) + ber[g * EPG + e];
    }

  // group softmax (wave-uniform after reductions; all lanes redundantly)
  float gm = glog[0];
  for (int g = 1; g < NGRP; ++g) gm = fmaxf(gm, glog[g]);
  float gp[NGRP], gs = 0.0f;
  for (int g = 0; g < NGRP; ++g) { gp[g] = expf(glog[g] - gm); gs += gp[g]; }
  for (int g = 0; g < NGRP; ++g) gp[g] /= gs;

  // top-2 groups (first-wins on ties, matching top_k)
  int a0 = 0;
  for (int g = 1; g < NGRP; ++g) if (gp[g] > gp[a0]) a0 = g;
  int a1 = (a0 == 0) ? 1 : 0;
  for (int g = 0; g < NGRP; ++g) if (g != a0 && gp[g] > gp[a1]) a1 = g;

  const float cf = conf[n];
  float p8[8]; int id8[8];
  const int gsel[2] = {a0, a1};
  for (int s = 0; s < 2; ++s) {
    int g = gsel[s];
    float m = elog[g][0];
    for (int e = 1; e < EPG; ++e) m = fmaxf(m, elog[g][e]);
    float sum = 0.0f, sp[EPG];
    for (int e = 0; e < EPG; ++e) { sp[e] = expf(elog[g][e] - m); sum += sp[e]; }
    for (int e = 0; e < EPG; ++e) {
      p8[s * EPG + e]  = (sp[e] / sum) * cf * gp[g];
      id8[s * EPG + e] = g * EPG + e;
    }
  }
  float tot = 1e-9f;
  for (int i = 0; i < 8; ++i) tot += p8[i];
  for (int i = 0; i < 8; ++i) p8[i] /= tot;

  // top-4 of the 8 candidates, renormalize
  bool used[8] = {false, false, false, false, false, false, false, false};
  int pick[4]; float ssum = 0.0f;
  for (int t = 0; t < 4; ++t) {
    int best = -1;
    for (int i = 0; i < 8; ++i)
      if (!used[i] && (best < 0 || p8[i] > p8[best])) best = i;
    used[best] = true; pick[t] = best; ssum += p8[best];
  }
  if (lane < NEXP) {
    float val = 0.0f;
    for (int t = 0; t < 4; ++t)
      if (id8[pick[t]] == lane) val = p8[pick[t]] / ssum;
    cw[(size_t)n * NEXP + lane] = val;
  }
}

// ---------------- kernel 4: build per-expert token lists ---------------------
__global__ __launch_bounds__(256) void dispatch_kernel(
    const float* __restrict__ cw, int* __restrict__ counts,
    int* __restrict__ tlist, float* __restrict__ wlist) {
  int n = blockIdx.x * blockDim.x + threadIdx.x;
  if (n >= NTOK) return;
  for (int e = 0; e < NEXP; ++e) {
    float w = cw[(size_t)n * NEXP + e];
    if (w > 0.0f) {
      int slot = atomicAdd(&counts[e], 1);
      tlist[e * NTOK + slot] = n;
      wlist[e * NTOK + slot] = w;
    }
  }
}

// ---------------- kernel 5: fused gathered expert FFN ------------------------
// out[tok,:] += w * (GELU(x[tok,:] @ W1[e] + b1[e]) @ W2[e] + b2[e])
// x rows are gathered into LDS by the Tensor Data Mover (gather-mode TDM),
// 16 row-indices per descriptor, tracked by TENSORcnt.
__global__ __launch_bounds__(256) void expert_ffn_kernel(
    const float* __restrict__ x,
    const float* __restrict__ W1, const float* __restrict__ b1,
    const float* __restrict__ W2, const float* __restrict__ b2,
    const int* __restrict__ counts, const int* __restrict__ tlist,
    const float* __restrict__ wlist, float* __restrict__ out) {
  const int e = blockIdx.y;
  const int cnt = counts[e];
  const int row0 = blockIdx.x * TM;
  if (row0 >= cnt) return;

  const int tid = threadIdx.x, wave = tid >> 5, lane = tid & 31;
  const int lm = lane & 15;
  const int ksel = (lane >> 4) * 8;   // K-half select; also C/D M-offset

  __shared__ float  xtile[TM][DIM];   // gathered fp32 token rows, 128 KB (TDM dest)
  __shared__ __bf16 hbuf[TM][HC];     // GELU'd hidden chunk, bf16, 4 KB
  __shared__ int    stok[TM];
  __shared__ float  swt[TM];

  if (tid < TM) {
    int r = row0 + tid;
    if (r < cnt) { stok[tid] = tlist[e * NTOK + r]; swt[tid] = wlist[e * NTOK + r]; }
    else         { stok[tid] = tlist[e * NTOK];     swt[tid] = 0.0f; }  // pad: weight 0
  }
  __syncthreads();

  // ---- TDM gather: two 16-row descriptors fill xtile from x ----
  if (wave == 0) {
    const unsigned long long gaddr = (unsigned long long)(uintptr_t)x;
    const unsigned int ga_lo = (unsigned int)(gaddr & 0xFFFFFFFFu);
    const unsigned int ga_hi = (unsigned int)((gaddr >> 32) & 0x1FFFFFFu);
    const unsigned int lds0  = (unsigned int)(uintptr_t)(&xtile[0][0]);
#pragma unroll
    for (int half = 0; half < 2; ++half) {
      // D# group 0: count=1 | gather_mode=1 (16-bit idx) | lds | global | type=2
      v4u g0;
      g0[0] = 0x80000001u;
      g0[1] = lds0 + (unsigned int)(half * 16 * DIM * 4);
      g0[2] = ga_lo;
      g0[3] = ga_hi | (2u << 30);
      // D# group 1: data_size=4B; tensor_dim0=DIM; tensor_dim1=NTOK;
      // tile_dim0=DIM; tile_dim1=16 indices; tensor_dim0_stride=DIM
      v8i g1;
      g1[0] = (2 << 16);
      g1[1] = (int)((unsigned)(DIM & 0xFFFF) << 16);
      g1[2] = (int)(((unsigned)DIM >> 16) | ((unsigned)(NTOK & 0xFFFF) << 16));
      g1[3] = (int)(((unsigned)NTOK >> 16) | ((unsigned)(DIM & 0xFFFF) << 16));
      g1[4] = 16;
      g1[5] = DIM;
      g1[6] = 0;
      g1[7] = 0;
      // D# groups 2-3: sixteen 16-bit row indices (token ids)
      const int b = half * 16;
      v4i g2, g3;
      g2[0] = stok[b + 0] | (stok[b + 1] << 16);
      g2[1] = stok[b + 2] | (stok[b + 3] << 16);
      g2[2] = stok[b + 4] | (stok[b + 5] << 16);
      g2[3] = stok[b + 6] | (stok[b + 7] << 16);
      g3[0] = stok[b + 8] | (stok[b + 9] << 16);
      g3[1] = stok[b + 10] | (stok[b + 11] << 16);
      g3[2] = stok[b + 12] | (stok[b + 13] << 16);
      g3[3] = stok[b + 14] | (stok[b + 15] << 16);
#if __clang_major__ >= 23
      v8i g4 = {0, 0, 0, 0, 0, 0, 0, 0};
      __builtin_amdgcn_tensor_load_to_lds(g0, g1, g2, g3, g4, 0);
#else
      __builtin_amdgcn_tensor_load_to_lds(g0, g1, g2, g3, 0);
#endif
    }
    __builtin_amdgcn_s_wait_tensorcnt(0);
  }
  __syncthreads();

  const float* W1e = W1 + (size_t)e * DIM * HID;
  const float* W2e = W2 + (size_t)e * HID * DIM;
  const float* b1e = b1 + (size_t)e * HID;
  const float* b2e = b2 + (size_t)e * DIM;

  const int rt1 = wave & 1, ct1 = wave >> 1;   // phase 1: 2x4 tiles of [32 x 64] h
  const int rt2 = wave >> 2, cg = wave & 3;    // phase 2: 2x(4x16) tiles of [32 x 1024]
  const int arow1 = rt1 * 16 + lm;             // this lane's row in xtile (phase 1)

  v8f acc2[16];
#pragma unroll
  for (int t = 0; t < 16; ++t)
#pragma unroll
    for (int i = 0; i < 8; ++i) acc2[t][i] = 0.0f;

  for (int hc = 0; hc < HID; hc += HC) {
    // ---- phase 1: h tile = GELU(x @ W1[:, hc:hc+64] + b1) ----
    v8f hacc;
#pragma unroll
    for (int i = 0; i < 8; ++i) hacc[i] = 0.0f;
    const int hcol = hc + ct1 * 16 + lm;
    for (int kb = 0; kb < DIM; kb += 32) {
      __builtin_prefetch(W1e + (size_t)(kb + 32 + ksel) * HID + hcol, 0, 1);
      v16bf a;
      {
        const float* p = &xtile[arow1][kb + ksel];   // LDS (ds_load) + cvt
#pragma unroll
        for (int j = 0; j < 8; ++j) { a[j] = (__bf16)p[j]; a[8 + j] = (__bf16)p[16 + j]; }
      }
      v16bf b = load_b_frag(W1e, HID, kb, ksel, hcol);
      hacc = __builtin_amdgcn_wmma_f32_16x16x32_bf16(false, a, false, b,
                                                     (short)0, hacc, false, false);
    }
    const float bias1 = b1e[hcol];
    __syncthreads();   // previous chunk's phase-2 readers are done
#pragma unroll
    for (int i = 0; i < 8; ++i) {
      float v = gelu_exact(hacc[i] + bias1);
      hbuf[rt1 * 16 + i + ksel][ct1 * 16 + lm] = (__bf16)v;
    }
    __syncthreads();
    // ---- phase 2: acc2 += h_tile @ W2[hc:hc+64, :] ----
    for (int ks = 0; ks < HC; ks += 32) {
      v16bf a;
      const __bf16* hp = &hbuf[rt2 * 16 + lm][ks + ksel];
#pragma unroll
      for (int j = 0; j < 8; ++j) { a[j] = hp[j]; a[8 + j] = hp[16 + j]; }
#pragma unroll
      for (int t = 0; t < 16; ++t) {
        int nb = (cg * 16 + t) * 16 + lm;
        __builtin_prefetch(W2e + (size_t)(hc + ks + 32 + ksel) * DIM + nb, 0, 1);
        v16bf b = load_b_frag(W2e, DIM, hc + ks, ksel, nb);
        acc2[t] = __builtin_amdgcn_wmma_f32_16x16x32_bf16(false, a, false, b,
                                                          (short)0, acc2[t], false, false);
      }
    }
  }

  // ---- epilogue: out[tok, c] += w * (acc + b2[c]) ----
#pragma unroll
  for (int i = 0; i < 8; ++i) {
    int r = rt2 * 16 + i + ksel;
    int tok = stok[r];
    float wt = swt[r];
    float* orow = out + (size_t)tok * DIM;
#pragma unroll
    for (int t = 0; t < 16; ++t) {
      int c = (cg * 16 + t) * 16 + lm;
      atomicAdd(orow + c, wt * (acc2[t][i] + b2e[c]));
    }
  }
}

// ---------------- host launcher ----------------------------------------------
extern "C" void kernel_launch(void* const* d_in, const int* in_sizes, int n_in,
                              void* d_out, int out_size, void* d_ws, size_t ws_size,
                              hipStream_t stream) {
  const float* x   = (const float*)d_in[0];
  const float* Wg  = (const float*)d_in[1];
  const float* bg  = (const float*)d_in[2];
  const float* Wer = (const float*)d_in[3];
  const float* ber = (const float*)d_in[4];
  const float* Wc1 = (const float*)d_in[5];
  const float* bc1 = (const float*)d_in[6];
  const float* Wc2 = (const float*)d_in[7];
  const float* bc2 = (const float*)d_in[8];
  const float* W1  = (const float*)d_in[9];
  const float* b1  = (const float*)d_in[10];
  const float* W2  = (const float*)d_in[11];
  const float* b2  = (const float*)d_in[12];
  float* out = (float*)d_out;

  // workspace carve-up (256B aligned)
  char* ws = (char*)d_ws;
  size_t off = 0;
  auto carve = [&](size_t bytes) -> void* {
    void* p = ws + off;
    off = (off + bytes + 255) & ~(size_t)255;
    return p;
  };
  float* C      = (float*)carve((size_t)NTOK * CHALF * sizeof(float)); // 4 MB
  float* conf   = (float*)carve((size_t)NTOK * sizeof(float));
  float* cw     = (float*)carve((size_t)NTOK * NEXP * sizeof(float));
  int*   counts = (int*)  carve((size_t)NEXP * sizeof(int));
  int*   tlist  = (int*)  carve((size_t)NEXP * NTOK * sizeof(int));
  float* wlist  = (float*)carve((size_t)NEXP * NTOK * sizeof(float));
  (void)ws_size; (void)in_sizes; (void)n_in;

  // 1) zero output (incl. aux_loss scalar at the tail) and counters
  zero_f32<<<(out_size + 255) / 256, 256, 0, stream>>>(out, out_size);
  zero_f32<<<1, 256, 0, stream>>>((float*)counts, NEXP);

  // 2) confidence estimator
  conf_hidden_kernel<<<dim3(NTOK / 32, CHALF / 64), 256, 0, stream>>>(x, Wc1, bc1, C);
  conf_out_kernel<<<NTOK / 8, 256, 0, stream>>>(C, Wc2, bc2, conf);

  // 3) hierarchical routing -> combine weights
  router_kernel<<<NTOK / 8, 256, 0, stream>>>(x, Wg, bg, Wer, ber, conf, cw);

  // 4) build gathered per-expert token lists
  dispatch_kernel<<<NTOK / 256, 256, 0, stream>>>(cw, counts, tlist, wlist);

  // 5) fused expert FFN (TDM-gathered, WMMA), scatter-add into out
  expert_ffn_kernel<<<dim3(NTOK / TM, NEXP), 256, 0, stream>>>(
      x, W1, b1, W2, b2, counts, tlist, wlist, out);
}
